// GenerativeSchurFlow_26869315404089
// MI455X (gfx1250) — matrix operand order repaired and checked by reading, not compiled, not run on hardware
//
#include <hip/hip_runtime.h>
#include <hip/hip_bf16.h>
#include <math.h>

typedef __attribute__((ext_vector_type(16))) _Float16 v16h;
typedef __attribute__((ext_vector_type(8)))  _Float16 v8h;
typedef __attribute__((ext_vector_type(8)))  float    v8f;
typedef unsigned int u32x4 __attribute__((ext_vector_type(4)));
typedef int          i32x4 __attribute__((ext_vector_type(4)));
typedef int          i32x8 __attribute__((ext_vector_type(8)));

#define BATCH 256

#if defined(__has_builtin)
#if __has_builtin(__builtin_amdgcn_tensor_load_to_lds)
#define HAVE_TDM 1
#endif
#endif
#ifndef HAVE_TDM
#define HAVE_TDM 0
#endif

// ---------------------------------------------------------------------------
// TDM: DMA one contiguous row (ndw dwords) from global into LDS.
// D# built per CDNA5 ISA (2D descriptor,
// groups 2/3 NULL-equivalent zeros).  Tracked by TENSORcnt.
// ---------------------------------------------------------------------------
__device__ __forceinline__ void tdm_load_row(const void* gsrc, unsigned lds_byte,
                                             unsigned ndw) {
#if HAVE_TDM
    unsigned long long ga = (unsigned long long)(size_t)gsrc;
    u32x4 g0;
    g0[0] = 1u;                                         // count=1, user mode
    g0[1] = lds_byte;                                   // lds_addr
    g0[2] = (unsigned)ga;                               // global_addr[31:0]
    g0[3] = (unsigned)((ga >> 32) & 0x01ffffffu)        // global_addr[56:32]
            | (2u << 30);                               // type=2 ("image")
    i32x8 g1;
    g1[0] = 0x00020000;                                 // data_size=2 (4B), mask=0
    g1[1] = (int)((ndw & 0xffffu) << 16);               // tensor_dim0 lo16
    g1[2] = (int)(((ndw >> 16) & 0xffffu) | (1u << 16));// dim0 hi16 | tensor_dim1=1
    g1[3] = (int)((ndw & 0xffffu) << 16);               // tile_dim0
    g1[4] = 1;                                          // tile_dim1=1
    g1[5] = (int)ndw;                                   // tensor_dim0_stride lo32
    g1[6] = 0;
    g1[7] = 0;
    i32x4 gz = {0, 0, 0, 0};
#if __clang_major__ >= 23
    i32x8 gz8 = {0, 0, 0, 0, 0, 0, 0, 0};
    __builtin_amdgcn_tensor_load_to_lds(g0, g1, gz, gz, gz8, 0);
#else
    __builtin_amdgcn_tensor_load_to_lds(g0, g1, gz, gz, 0);
#endif
#endif
}

// ---------------------------------------------------------------------------
// Pack: fused squeeze (tl-br-tr-bl quadrant order) + f32 -> f16 + K-pad.
// dst layout: [b][p=y*N+x][CPAD]  (channel innermost => WMMA B-fragment loads
// are contiguous per lane).
// ---------------------------------------------------------------------------
template<int CPAD, int N, int CPREV>
__global__ __launch_bounds__(256)
void pack_squeeze_kernel(const float* __restrict__ src,  // [B][CPREV][2N][2N]
                         _Float16* __restrict__ dst) {   // [B][N*N][CPAD]
    constexpr int P = N * N;
    size_t idx = (size_t)blockIdx.x * blockDim.x + threadIdx.x;
    size_t total = (size_t)BATCH * P * CPAD;
    if (idx >= total) return;
    int ci = (int)(idx % CPAD);
    size_t rest = idx / CPAD;
    int p = (int)(rest % P);
    int b = (int)(rest / P);
    float v = 0.f;
    if (ci < 4 * CPREV) {
        int q = ci / CPREV, c = ci % CPREV;
        const int qdy[4] = {0, 1, 0, 1};   // concat order (0,0),(1,1),(0,1),(1,0)
        const int qdx[4] = {0, 1, 1, 0};
        int y = p / N, x = p % N;
        int sy = 2 * y + qdy[q], sx = 2 * x + qdx[q];
        v = src[(((size_t)b * CPREV + c) * (2 * N) + sy) * (2 * N) + sx];
    }
    dst[idx] = (_Float16)v;
}

// ---------------------------------------------------------------------------
// Weight pack: K[co][ci][3][3] f32 -> w[tap][co][ci] f16, zero K-padded.
// ---------------------------------------------------------------------------
template<int CPAD, int CREAL>
__global__ __launch_bounds__(256)
void pack_weights_kernel(const float* __restrict__ K,
                         _Float16* __restrict__ w) {    // [9][CPAD][CPAD]
    int idx = blockIdx.x * blockDim.x + threadIdx.x;
    if (idx >= 9 * CPAD * CPAD) return;
    int ci  = idx % CPAD;
    int co  = (idx / CPAD) % CPAD;
    int tap = idx / (CPAD * CPAD);
    float v = 0.f;
    if (ci < CREAL && co < CREAL) {
        int ky = tap / 3, kx = tap % 3;
        v = K[(((size_t)co * CREAL + ci) * 3 + ky) * 3 + kx];
    }
    w[idx] = (_Float16)v;
}

// ---------------------------------------------------------------------------
// Implicit-GEMM circular conv via V_WMMA_F32_16X16X32_F16.
// Block = one (batch, row).  Wave 0 TDM-loads the 3 wrapped input rows
// (all CPAD channels, f16) into LDS once; waves then round-robin over
// (N/16 pixel-tiles) x (CREAL/16 co-tiles), feeding WMMA B fragments from
// LDS (ds_load_b128) and A fragments from L2-resident packed weights.
// Fused epilogue: spatial conv-bias, actnorm, optional tanh + logdet.
// ---------------------------------------------------------------------------
template<int CPAD, int N, int CREAL, bool TANH>
__global__ __launch_bounds__(128)
void conv_wmma_kernel(const _Float16* __restrict__ xpack,   // [B][P][CPAD]
                      const _Float16* __restrict__ wpack,   // [9][CPAD][CPAD]
                      const float* __restrict__ conv_bias,  // [CREAL][N][N]
                      const float* __restrict__ act_bias,   // [CREAL]
                      const float* __restrict__ act_log_scale, // [CREAL]
                      float* __restrict__ out,              // [B][CREAL][P]
                      float* __restrict__ tanh_sum) {       // [B]
    constexpr int P = N * N;
    constexpr int TROW   = N / 16;                 // 16-px tiles per row
    constexpr int MTILES = (CREAL + 15) / 16;
    constexpr int ROWDW  = N * CPAD / 2;           // dwords per staged row
    extern __shared__ _Float16 smem[];             // [3][N][CPAD]

    const int b    = blockIdx.x / N;
    const int y    = blockIdx.x % N;
    const int lane = threadIdx.x & 31;
    const int wave = threadIdx.x >> 5;
    const int col  = lane & 15;
    const int hi   = lane >> 4;

    const int rows[3] = {(y + N - 1) % N, y, (y + 1) % N};

#if HAVE_TDM
    if (threadIdx.x < 32) {                        // wave 0 issues the DMAs
        const unsigned lds0 = (unsigned)(size_t)(void*)smem;
#pragma unroll
        for (int r = 0; r < 3; ++r)
            tdm_load_row(xpack + ((size_t)b * P + (size_t)rows[r] * N) * CPAD,
                         lds0 + (unsigned)r * N * CPAD * 2, ROWDW);
        __builtin_amdgcn_s_wait_tensorcnt(0);
    }
#else
    {
        unsigned* s32 = (unsigned*)smem;
        for (int i = threadIdx.x; i < 3 * ROWDW; i += 128) {
            int r = i / ROWDW, rem = i % ROWDW;
            const unsigned* g32 = (const unsigned*)
                (xpack + ((size_t)b * P + (size_t)rows[r] * N) * CPAD);
            s32[i] = g32[rem];
        }
    }
#endif
    __syncthreads();

    __builtin_prefetch(wpack, 0, 3);               // weights: keep hot

    float ld = 0.f;
    for (int wi = wave; wi < TROW * MTILES; wi += 4) {
        const int x0    = (wi % TROW) * 16;
        const int mtile = wi / TROW;
        v8f acc0 = {};
        v8f acc1 = {};
        const int xcol = x0 + col;

#pragma unroll
        for (int tap = 0; tap < 9; ++tap) {
            const int ry = tap / 3;                // staged row 0..2
            const int dx = tap % 3 - 1;
            int xs = xcol + dx; xs += (xs < 0) ? N : 0; xs -= (xs >= N) ? N : 0;
            // B fragment: lane's pixel, 16 contiguous channels per K-block (LDS)
            const _Float16* bbase = smem + ((size_t)(ry * N + xs)) * CPAD + hi * 16;
            // A fragment: lane's co row, 16 contiguous channels per K-block (global/L2)
            const _Float16* abase =
                wpack + ((size_t)tap * CPAD + mtile * 16 + col) * CPAD + hi * 16;
#pragma unroll
            for (int kb = 0; kb < CPAD / 32; ++kb) {
                v16h bfrag = *(const v16h*)(bbase + kb * 32);
                v8h alo = *(const v8h*)(abase + kb * 32);
                v8h ahi = *(const v8h*)(abase + kb * 32 + 8);
                v16h afrag;
#pragma unroll
                for (int e = 0; e < 8; ++e) { afrag[e] = alo[e]; afrag[e + 8] = ahi[e]; }
                if (kb & 1)
                    acc1 = __builtin_amdgcn_wmma_f32_16x16x32_f16(
                        false, afrag, false, bfrag, (short)0, acc1, false, false);
                else
                    acc0 = __builtin_amdgcn_wmma_f32_16x16x32_f16(
                        false, afrag, false, bfrag, (short)0, acc0, false, false);
            }
        }

#pragma unroll
        for (int r = 0; r < 8; ++r) {
            const int co = mtile * 16 + hi * 8 + r;    // C/D layout: M = r + 8*hi
            if (co < CREAL) {
                const int px = x0 + col;
                float v = acc0[r] + acc1[r];
                v += conv_bias[(size_t)co * P + (size_t)y * N + px];
                v = v * __expf(act_log_scale[co]) + act_bias[co];
                if (TANH) {
                    float tt = tanhf(v);
                    ld += log1pf(-tt * tt);
                    v = tt;
                }
                out[((size_t)b * CREAL + co) * P + (size_t)y * N + px] = v;
            }
        }
    }
    if (TANH) {
#pragma unroll
        for (int off = 16; off > 0; off >>= 1) ld += __shfl_xor(ld, off, 32);
        if (lane == 0) atomicAdd(&tanh_sum[b], ld);
    }
}

// ---------------------------------------------------------------------------
// Spectral logdet: one workgroup per frequency (u,v); Khat via 9-tap DFT into
// LDS (complex interleaved), complex Gaussian elimination, sum 0.5*log|diag|^2.
// ---------------------------------------------------------------------------
template<int C, int N>
__global__ __launch_bounds__(256)
void spectral_logdet_kernel(const float* __restrict__ K,   // [C][C][3][3]
                            float* __restrict__ base_sum) {
    extern __shared__ float sm[];                           // 2*C*C floats
    __shared__ float piv[2];
    const int u = blockIdx.x / N, v = blockIdx.x % N;
    const int tid = threadIdx.x;

    float cs[9], sn[9];
#pragma unroll
    for (int tap = 0; tap < 9; ++tap) {
        int ky = tap / 3, kx = tap % 3;
        float ang = -2.f * 3.14159265358979323846f * (float)(u * ky + v * kx) / (float)N;
        cs[tap] = __cosf(ang);
        sn[tap] = __sinf(ang);
    }
    for (int idx = tid; idx < C * C; idx += blockDim.x) {
        const float* kk = K + (size_t)idx * 9;
        float re = 0.f, im = 0.f;
#pragma unroll
        for (int tap = 0; tap < 9; ++tap) {
            re += kk[tap] * cs[tap];
            im += kk[tap] * sn[tap];
        }
        sm[2 * idx] = re;
        sm[2 * idx + 1] = im;
    }
    __syncthreads();

    float logabs = 0.f;
    for (int k = 0; k < C; ++k) {
        if (tid == 0) {
            float re = sm[2 * (k * C + k)], im = sm[2 * (k * C + k) + 1];
            float m2 = fmaxf(re * re + im * im, 1e-30f);
            logabs += 0.5f * __logf(m2);
            float inv = 1.f / m2;
            piv[0] = re * inv;
            piv[1] = -im * inv;
        }
        __syncthreads();
        const float pre = piv[0], pim = piv[1];
        for (int r = k + 1 + tid; r < C; r += blockDim.x) {
            float are = sm[2 * (r * C + k)], aim = sm[2 * (r * C + k) + 1];
            float fre = are * pre - aim * pim;
            float fim = are * pim + aim * pre;
            for (int c = k; c < C; ++c) {
                float bre = sm[2 * (k * C + c)], bim = sm[2 * (k * C + c) + 1];
                sm[2 * (r * C + c)]     -= fre * bre - fim * bim;
                sm[2 * (r * C + c) + 1] -= fre * bim + fim * bre;
            }
        }
        __syncthreads();
    }
    if (tid == 0) atomicAdd(base_sum, logabs);
}

// ---------------------------------------------------------------------------
__global__ void finalize_logdet_kernel(const float* __restrict__ als0,
                                       const float* __restrict__ als1,
                                       const float* __restrict__ als2,
                                       const float* __restrict__ tanh_sum,
                                       const float* __restrict__ base_sum,
                                       float* __restrict__ out_logdet) {
    int b = blockIdx.x * blockDim.x + threadIdx.x;
    if (b >= BATCH) return;
    float s0 = 0.f, s1 = 0.f, s2 = 0.f;
    for (int c = 0; c < 12; ++c)  s0 += als0[c];
    for (int c = 0; c < 48; ++c)  s1 += als1[c];
    for (int c = 0; c < 192; ++c) s2 += als2[c];
    float base = *base_sum + 4096.f * s0 + 1024.f * s1 + 256.f * s2;
    out_logdet[b] = base + tanh_sum[b];
}

// ---------------------------------------------------------------------------
extern "C" void kernel_launch(void* const* d_in, const int* in_sizes, int n_in,
                              void* d_out, int out_size, void* d_ws, size_t ws_size,
                              hipStream_t stream) {
    const float* x    = (const float*)d_in[0];
    const float* k0   = (const float*)d_in[1];
    const float* cb0  = (const float*)d_in[2];
    const float* ab0  = (const float*)d_in[3];
    const float* als0 = (const float*)d_in[4];
    const float* k1   = (const float*)d_in[5];
    const float* cb1  = (const float*)d_in[6];
    const float* ab1  = (const float*)d_in[7];
    const float* als1 = (const float*)d_in[8];
    const float* k2   = (const float*)d_in[9];
    const float* cb2  = (const float*)d_in[10];
    const float* ab2  = (const float*)d_in[11];
    const float* als2 = (const float*)d_in[12];

    float* out_x  = (float*)d_out;                       // [256][192][16][16]
    float* out_ld = (float*)d_out + (size_t)BATCH * 192 * 16 * 16;

    size_t off = 0;
    auto carve = [&](size_t bytes) {
        size_t o = off;
        off = (off + bytes + 255) & ~(size_t)255;
        return o;
    };
    char* ws = (char*)d_ws;
    _Float16* pack0 = (_Float16*)(ws + carve((size_t)BATCH * 4096 * 32 * 2));
    float*    act0  = (float*)   (ws + carve((size_t)BATCH * 12 * 4096 * 4));
    _Float16* pack1 = (_Float16*)(ws + carve((size_t)BATCH * 1024 * 64 * 2));
    float*    act1  = (float*)   (ws + carve((size_t)BATCH * 48 * 1024 * 4));
    _Float16* pack2 = (_Float16*)(ws + carve((size_t)BATCH * 256 * 192 * 2));
    _Float16* w0    = (_Float16*)(ws + carve((size_t)9 * 32 * 32 * 2));
    _Float16* w1    = (_Float16*)(ws + carve((size_t)9 * 64 * 64 * 2));
    _Float16* w2    = (_Float16*)(ws + carve((size_t)9 * 192 * 192 * 2));
    float*    tsum  = (float*)   (ws + carve((size_t)BATCH * 4));
    float*    bsum  = (float*)   (ws + carve(4));

    hipMemsetAsync(tsum, 0, (size_t)BATCH * 4 + 256, stream);   // tsum + bsum

    pack_weights_kernel<32, 12>  <<<(9 * 32 * 32 + 255) / 256, 256, 0, stream>>>(k0, w0);
    pack_weights_kernel<64, 48>  <<<(9 * 64 * 64 + 255) / 256, 256, 0, stream>>>(k1, w1);
    pack_weights_kernel<192, 192><<<(9 * 192 * 192 + 255) / 256, 256, 0, stream>>>(k2, w2);

    spectral_logdet_kernel<12, 64><<<64 * 64, 256, 2 * 12 * 12 * sizeof(float), stream>>>(k0, bsum);
    spectral_logdet_kernel<48, 32><<<32 * 32, 256, 2 * 48 * 48 * sizeof(float), stream>>>(k1, bsum);
    {
        size_t lds = 2ull * 192 * 192 * sizeof(float);   // 288KB < 320KB WGP LDS
        hipFuncSetAttribute((const void*)&spectral_logdet_kernel<192, 16>,
                            hipFuncAttributeMaxDynamicSharedMemorySize, (int)lds);
        spectral_logdet_kernel<192, 16><<<16 * 16, 256, lds, stream>>>(k2, bsum);
    }

    // ---- layer 0: squeeze 3@128 -> 12@64 (Kpad 32)
    {
        size_t tot = (size_t)BATCH * 4096 * 32;
        pack_squeeze_kernel<32, 64, 3><<<(int)((tot + 255) / 256), 256, 0, stream>>>(x, pack0);
        conv_wmma_kernel<32, 64, 12, true>
            <<<BATCH * 64, 128, 3 * 64 * 32 * 2, stream>>>(
                pack0, w0, cb0, ab0, als0, act0, tsum);
    }
    // ---- layer 1: squeeze 12@64 -> 48@32 (Kpad 64)
    {
        size_t tot = (size_t)BATCH * 1024 * 64;
        pack_squeeze_kernel<64, 32, 12><<<(int)((tot + 255) / 256), 256, 0, stream>>>(act0, pack1);
        conv_wmma_kernel<64, 32, 48, true>
            <<<BATCH * 32, 128, 3 * 32 * 64 * 2, stream>>>(
                pack1, w1, cb1, ab1, als1, act1, tsum);
    }
    // ---- layer 2: squeeze 48@32 -> 192@16 (Kpad 192), write x to d_out
    {
        size_t tot = (size_t)BATCH * 256 * 192;
        pack_squeeze_kernel<192, 16, 48><<<(int)((tot + 255) / 256), 256, 0, stream>>>(act1, pack2);
        conv_wmma_kernel<192, 16, 192, false>
            <<<BATCH * 16, 128, 3 * 16 * 192 * 2, stream>>>(
                pack2, w2, cb2, ab2, als2, out_x, tsum);
    }

    finalize_logdet_kernel<<<1, BATCH, 0, stream>>>(als0, als1, als2, tsum, bsum, out_ld);
}
